// WindowAttention_4853313044771
// MI455X (gfx1250) — matrix operand order repaired and checked by reading, not compiled, not run on hardware
//
#include <hip/hip_runtime.h>
#include <hip/hip_bf16.h>
#include <hip/hip_fp16.h>

#define DIM     384
#define HEADS   12
#define HD      32
#define NTOK    49
#define NPAD    64
#define NWIN    64

typedef __attribute__((ext_vector_type(16))) _Float16 v16h;
typedef __attribute__((ext_vector_type(8)))  float    v8f;

// ---- LDS strides (in halves); chosen so byte stride % 16 == 0 (b128-aligned)
// and dword-stride % 64 spreads rows across banks.
#define SX_STRIDE 392   // 784 B
#define SQ_STRIDE 40    // 80 B
#define SV_STRIDE 72    // 144 B
#define SP_STRIDE 72    // 144 B

// ---------------- fragment loaders ----------------
// A-matrix 16x32 f16: lane L holds row m = L%16, g = L/16;
// halves 0..15 = K (g*16 + 0..15), matching frag_b.
__device__ __forceinline__ v16h frag_a(const _Float16* base, int row, int strideH,
                                       int kOffH, int g) {
    union { uint4 u[2]; v16h h; } f;
    const _Float16* p = base + row * strideH + kOffH + g * 16;
    f.u[0] = *(const uint4*)(p);
    f.u[1] = *(const uint4*)(p + 8);
    return f.h;
}

// B-matrix 32x16 f16: lane L holds column n = L%16, K = (L/16)*16 + 0..15,
// i.e. 16 contiguous halves at p (caller folds column/row addressing in).
__device__ __forceinline__ v16h frag_b(const _Float16* p) {
    union { uint4 u[2]; v16h h; } f;
    f.u[0] = *(const uint4*)(p);
    f.u[1] = *(const uint4*)(p + 8);
    return f.h;
}

__device__ __forceinline__ v16h zero_frag() {
    union { uint4 u[2]; v16h h; } f;
    f.u[0] = make_uint4(0u, 0u, 0u, 0u);
    f.u[1] = make_uint4(0u, 0u, 0u, 0u);
    return f.h;
}

#define WMMA_F16(A, B, C) \
    __builtin_amdgcn_wmma_f32_16x16x32_f16(false, (A), false, (B), (short)0, (C), false, false)

// ---------------- prep kernels ----------------
__global__ void prep_weights(const float* __restrict__ qkv_w,
                             const float* __restrict__ proj_w,
                             _Float16* __restrict__ qw16,
                             _Float16* __restrict__ pw16) {
    int i = blockIdx.x * 256 + threadIdx.x;
    if (i < 3 * DIM * DIM) qw16[i] = (_Float16)qkv_w[i];
    if (i < DIM * DIM)     pw16[i] = (_Float16)proj_w[i];
}

// combined bias+mask, padded to 64x64 with -1e30 on invalid columns/rows:
// comb[win][h][i][j]
__global__ void prep_comb(const float* __restrict__ mask,
                          const float* __restrict__ rpb,
                          const int*   __restrict__ rel_idx,
                          float* __restrict__ comb) {
    int i = blockIdx.x * 256 + threadIdx.x;
    if (i >= NWIN * HEADS * NPAD * NPAD) return;
    int j = i & 63;
    int r = (i >> 6) & 63;
    int h = (i >> 12) % HEADS;
    int w = i / (HEADS * NPAD * NPAD);
    float v = -1e30f;
    if (r < NTOK && j < NTOK) {
        v = rpb[rel_idx[r * NTOK + j] * HEADS + h] + mask[(w * NTOK + r) * NTOK + j];
    }
    comb[i] = v;
}

// ---------------- fused window-attention kernel ----------------
__global__ __launch_bounds__(256) void win_attn(
        const float*    __restrict__ x,
        const _Float16* __restrict__ qw16,
        const float*    __restrict__ qkv_b,
        const _Float16* __restrict__ pw16,
        const float*    __restrict__ proj_b,
        const float*    __restrict__ comb,
        float*          __restrict__ out) {
    const int b    = blockIdx.x;
    const int tid  = threadIdx.x;
    const int wave = tid >> 5;
    const int lane = tid & 31;
    const int g    = lane >> 4;   // half-wave group
    const int ln   = lane & 15;
    const int win  = b & (NWIN - 1);

    __shared__ _Float16 sX[NTOK * SX_STRIDE];   // x in f16, 38416 B
    __shared__ _Float16 sQ[NPAD * SQ_STRIDE];   // Q (scale folded), 5120 B
    __shared__ _Float16 sK[NPAD * SQ_STRIDE];   // K, 5120 B
    __shared__ _Float16 sVt[HD * SV_STRIDE];    // V transposed, 4608 B
    __shared__ _Float16 sP[NPAD * SP_STRIDE];   // probs; cols 0..31 reused for o_h, 9216 B

    // ---- stage x -> LDS f16 (vectorized) ----
    const float* xb = x + (size_t)b * NTOK * DIM;
    for (int idx = tid; idx < NTOK * DIM / 4; idx += 256) {
        float4 f = ((const float4*)xb)[idx];
        int e   = idx * 4;
        int row = e / DIM, col = e % DIM;
        _Float16* d = &sX[row * SX_STRIDE + col];
        d[0] = (_Float16)f.x; d[1] = (_Float16)f.y;
        d[2] = (_Float16)f.z; d[3] = (_Float16)f.w;
    }
    __syncthreads();

    // persistent projection accumulators: wave owns M-tile (wave&3),
    // N-tiles (wave>>2)*12 .. +11
    const int pmt    = wave & 3;
    const int pnbase = (wave >> 2) * 12;
    v8f pacc[12];
    for (int j = 0; j < 12; ++j) pacc[j] = (v8f){};

    const float scale = 0.17677669529663687f;   // HD^-0.5

    for (int h = 0; h < HEADS; ++h) {
        // ======== Phase A: Q/K/V for this head (all 8 waves, 24 tiles) ========
        for (int j = 0; j < 3; ++j) {
            int t   = wave + 8 * j;       // 0..23
            int mat = t >> 3;             // 0=Q 1=K 2=V
            int sub = t & 7;
            int mt  = sub >> 1;
            int nt  = sub & 1;
            int orow = mat * DIM + h * HD + nt * 16 + ln;   // weight output-channel
            const _Float16* wrow = qw16 + (size_t)orow * DIM;
            v8f acc = (v8f){};
            int arow = mt * 16 + ln;
            for (int kb = 0; kb < 12; ++kb) {
                v16h a = (arow < NTOK) ? frag_a(sX, arow, SX_STRIDE, kb * 32, g)
                                       : zero_frag();
                v16h bm = frag_b(wrow + kb * 32 + g * 16);
                acc = WMMA_F16(a, bm, acc);
            }
            float bias = qkv_b[orow];
            if (mat == 0) {
                for (int r = 0; r < 8; ++r)
                    sQ[(mt * 16 + r + 8 * g) * SQ_STRIDE + nt * 16 + ln] =
                        (_Float16)((acc[r] + bias) * scale);
            } else if (mat == 1) {
                for (int r = 0; r < 8; ++r)
                    sK[(mt * 16 + r + 8 * g) * SQ_STRIDE + nt * 16 + ln] =
                        (_Float16)(acc[r] + bias);
            } else {   // V stored transposed: sVt[channel][token]
                for (int r = 0; r < 8; ++r)
                    sVt[(nt * 16 + ln) * SV_STRIDE + mt * 16 + r + 8 * g] =
                        (_Float16)(acc[r] + bias);
            }
        }
        __syncthreads();

        // ======== Phase B: scores + softmax + P@V (waves 0..3, EXEC uniform) ========
        if (wave < 4) {
            const int mt = wave;
            const float* combh = comb + ((size_t)win * HEADS + h) * (NPAD * NPAD);
            v8f s[4];
            v16h aq = frag_a(sQ, mt * 16 + ln, SQ_STRIDE, 0, g);
            for (int nt = 0; nt < 4; ++nt) {
                v16h bk = frag_b(&sK[(nt * 16 + ln) * SQ_STRIDE + g * 16]);
                v8f z = (v8f){};
                s[nt] = WMMA_F16(aq, bk, z);
                for (int r = 0; r < 8; ++r)
                    s[nt][r] += combh[(mt * 16 + r + 8 * g) * NPAD + nt * 16 + ln];
            }
            // softmax per row (row lives in 16 lanes of one half-group)
            for (int r = 0; r < 8; ++r) {
                float mx = fmaxf(fmaxf(s[0][r], s[1][r]), fmaxf(s[2][r], s[3][r]));
                for (int m = 1; m < 16; m <<= 1) mx = fmaxf(mx, __shfl_xor(mx, m, 32));
                float e0 = __expf(s[0][r] - mx), e1 = __expf(s[1][r] - mx);
                float e2 = __expf(s[2][r] - mx), e3 = __expf(s[3][r] - mx);
                float sum = e0 + e1 + e2 + e3;
                for (int m = 1; m < 16; m <<= 1) sum += __shfl_xor(sum, m, 32);
                float inv = 1.0f / sum;
                int row = mt * 16 + r + 8 * g;
                sP[row * SP_STRIDE +  0 + ln] = (_Float16)(e0 * inv);
                sP[row * SP_STRIDE + 16 + ln] = (_Float16)(e1 * inv);
                sP[row * SP_STRIDE + 32 + ln] = (_Float16)(e2 * inv);
                sP[row * SP_STRIDE + 48 + ln] = (_Float16)(e3 * inv);
            }
            // o_h = P @ V   (A from sP rows of this tile; B from transposed V)
            v8f o0 = (v8f){}, o1 = (v8f){};
            for (int kb = 0; kb < 2; ++kb) {
                v16h ap = frag_a(sP, mt * 16 + ln, SP_STRIDE, kb * 32, g);
                v16h b0 = frag_b(&sVt[(ln)      * SV_STRIDE + kb * 32 + g * 16]);
                v16h b1 = frag_b(&sVt[(16 + ln) * SV_STRIDE + kb * 32 + g * 16]);
                o0 = WMMA_F16(ap, b0, o0);
                o1 = WMMA_F16(ap, b1, o1);
            }
            // overwrite consumed P rows (cols 0..31) with o_h
            for (int r = 0; r < 8; ++r) {
                int row = mt * 16 + r + 8 * g;
                sP[row * SP_STRIDE + ln]      = (_Float16)o0[r];
                sP[row * SP_STRIDE + 16 + ln] = (_Float16)o1[r];
            }
        }
        __syncthreads();

        // ======== Phase C: projection accumulation (all 8 waves) ========
        {
            v16h ao = frag_a(sP, pmt * 16 + ln, SP_STRIDE, 0, g);
            for (int j = 0; j < 12; ++j) {
                int oc = (pnbase + j) * 16 + ln;
                v16h bw = frag_b(pw16 + (size_t)oc * DIM + h * HD + g * 16);
                pacc[j] = WMMA_F16(ao, bw, pacc[j]);
            }
        }
        __syncthreads();   // before next head overwrites sQ/sK/sVt/sP
    }

    // ======== epilogue: add proj bias, store valid rows ========
    float* ob = out + (size_t)b * NTOK * DIM;
    for (int j = 0; j < 12; ++j) {
        int oc = (pnbase + j) * 16 + ln;
        float pb = proj_b[oc];
        for (int r = 0; r < 8; ++r) {
            int row = pmt * 16 + r + 8 * g;
            if (row < NTOK) ob[row * DIM + oc] = pacc[j][r] + pb;
        }
    }
}

// ---------------- host launcher ----------------
extern "C" void kernel_launch(void* const* d_in, const int* in_sizes, int n_in,
                              void* d_out, int out_size, void* d_ws, size_t ws_size,
                              hipStream_t stream) {
    (void)in_sizes; (void)n_in; (void)out_size; (void)ws_size;
    const float* x      = (const float*)d_in[0];
    const float* mask   = (const float*)d_in[1];
    const float* qkv_w  = (const float*)d_in[2];
    const float* qkv_b  = (const float*)d_in[3];
    const float* proj_w = (const float*)d_in[4];
    const float* proj_b = (const float*)d_in[5];
    const float* rpb    = (const float*)d_in[6];
    const int*   relidx = (const int*)d_in[7];

    char* ws = (char*)d_ws;
    _Float16* qw16 = (_Float16*)(ws);                       // 884736 B
    _Float16* pw16 = (_Float16*)(ws + 884736);              // 294912 B
    float*    comb = (float*)(ws + 884736 + 294912);        // 12.6 MB

    prep_weights<<<(3 * DIM * DIM + 255) / 256, 256, 0, stream>>>(qkv_w, proj_w, qw16, pw16);
    prep_comb<<<(NWIN * HEADS * NPAD * NPAD + 255) / 256, 256, 0, stream>>>(mask, rpb, relidx, comb);
    win_attn<<<4096, 256, 0, stream>>>(x, qw16, qkv_b, pw16, proj_b, comb, (float*)d_out);
}
